// EI_RNN_64381559767339
// MI455X (gfx1250) — matrix operand order repaired and checked by reading, not compiled, not run on hardware
//
#include <hip/hip_runtime.h>
#include <hip/hip_bf16.h>
#include <cmath>

typedef __attribute__((ext_vector_type(16))) _Float16 v16h;
typedef __attribute__((ext_vector_type(8)))  float    v8f;

#define B_  256
#define T_  1000
#define I_  128
#define H_  256
#define O_  64
#define BT  16          // batch rows per workgroup
#define NTH 512         // 16 wave32 waves per workgroup
#define ALPHA 0.1f      // DT/TAU = 10/100

// ---- CDNA5 async global->LDS copy (ASYNCcnt-tracked), inline-asm path ----
__device__ __forceinline__ void async_copy_b128(unsigned lds_off, const float* gaddr) {
  asm volatile("global_load_async_to_lds_b128 %0, %1, off"
               :: "v"(lds_off), "v"((unsigned long long)(uintptr_t)gaddr)
               : "memory");
}
__device__ __forceinline__ void wait_async0() {
  asm volatile("s_wait_asynccnt 0" ::: "memory");
}

// A operand: 16x32 (MxK) f16, ISA layout: lane<16 -> M=lane, halfs 0..7 = K k0..k0+7,
// halfs 8..15 = K k0+16..k0+23; lanes 16..31 hold the +8 K groups.
__device__ __forceinline__ v16h load_matA(const _Float16* base, int stride, int kbase, int lane) {
  const int m = lane & 15, g = lane >> 4;
  const _Float16* p = base + m * stride + kbase + 8 * g;
  v16h a;
#pragma unroll
  for (int i = 0; i < 8; ++i) { a[i] = p[i]; a[i + 8] = p[i + 16]; }
  return a;
}

// B operand: 32x16 (KxN) f16, ISA layout: lane<16 -> N=lane holds K k0..k0+7, k0+16..k0+23,
// lanes 16..31 hold the +8 K groups.  B[k][n] = W[n][k] (row-major weight rows).
__device__ __forceinline__ v16h load_matB(const _Float16* base, int stride, int n0, int kbase, int lane) {
  const int n = lane & 15, g = lane >> 4;
  const _Float16* p = base + (n0 + n) * stride + kbase + 8 * g;
  v16h b;
#pragma unroll
  for (int i = 0; i < 8; ++i) { b[i] = p[i]; b[i + 8] = p[i + 16]; }
  return b;
}

// Prefetch step `tstep` activations/noise into the selected raw LDS buffers.
// Wave w owns batch row (b0+w); lane covers 4 contiguous floats -> one b128 per lane,
// one async instruction covers a full 128-float row segment per wave.
__device__ __forceinline__ void prefetch_step(int tstep, int b0, int wave, int lane,
                                              const float* inputs, const float* in_noise,
                                              const float* re_noise,
                                              float* in_raw, float* nz_raw, float* re_raw) {
  const int c4 = lane * 4;
  const long ib = (long)(b0 + wave) * T_ * I_ + (long)tstep * I_ + c4;
  const long rb = (long)(b0 + wave) * T_ * H_ + (long)tstep * H_ + c4;
  async_copy_b128((unsigned)(uintptr_t)(in_raw + wave * I_ + c4), inputs + ib);
  async_copy_b128((unsigned)(uintptr_t)(nz_raw + wave * I_ + c4), in_noise + ib);
  async_copy_b128((unsigned)(uintptr_t)(re_raw + wave * H_ + c4), re_noise + rb);
  async_copy_b128((unsigned)(uintptr_t)(re_raw + wave * H_ + 128 + c4), re_noise + rb + 128);
}

__global__ __launch_bounds__(NTH, 1)
void ei_rnn_kernel(const float* __restrict__ inputs, const float* __restrict__ hidden,
                   const float* __restrict__ hh, const float* __restrict__ mask,
                   const float* __restrict__ W_ih, const float* __restrict__ bias,
                   const float* __restrict__ W_ho, const float* __restrict__ b_ho,
                   const float* __restrict__ in_noise, const float* __restrict__ re_noise,
                   float* __restrict__ h_final, float* __restrict__ outs)
{
  extern __shared__ char smem[];
  _Float16* whh_s  = (_Float16*)smem;                 // [H][H]  128 KB
  _Float16* wih_s  = whh_s + H_ * H_;                 // [H][I]   64 KB
  _Float16* who_s  = wih_s + H_ * I_;                 // [O][H]   32 KB
  float*    h_s    = (float*)(who_s + O_ * H_);       // [BT][H]  16 KB
  _Float16* r_s    = (_Float16*)(h_s + BT * H_);      // [BT][H]   8 KB
  _Float16* x_s    = r_s + BT * H_;                   // [BT][I]   4 KB
  float*    bias_s = (float*)(x_s + BT * I_);         // [H]       1 KB
  float*    bho_s  = bias_s + H_;                     // [O]     256 B
  float*    in_raw = bho_s + O_;                      // [2][BT*I] 16 KB (async dst)
  float*    nz_raw = in_raw + 2 * BT * I_;            // [2][BT*I] 16 KB (async dst)
  float*    re_raw = nz_raw + 2 * BT * I_;            // [2][BT*H] 32 KB (async dst)

  const int tid  = threadIdx.x;
  const int lane = tid & 31;
  const int wave = tid >> 5;            // 0..15, each owns one 16-col N-tile of H
  const int b0   = blockIdx.x * BT;

  // ---- kick off async prefetch of step 0 while we stage weights ----
  prefetch_step(0, b0, wave, lane, inputs, in_noise, re_noise, in_raw, nz_raw, re_raw);

  // ---- one-time: Dale's-law masked weights as f16, LDS-resident for all T steps ----
  for (int i = tid; i < H_ * H_; i += NTH)
    whh_s[i] = (_Float16)(fabsf(hh[i]) * mask[i]);
  for (int i = tid; i < H_ * I_; i += NTH)
    wih_s[i] = (_Float16)W_ih[i];
  for (int i = tid; i < O_ * H_; i += NTH)
    who_s[i] = (_Float16)W_ho[i];
  for (int i = tid; i < H_; i += NTH) bias_s[i] = bias[i];
  for (int i = tid; i < O_; i += NTH) bho_s[i] = b_ho[i];
  for (int i = tid; i < BT * H_; i += NTH) {
    const int m = i >> 8, n = i & (H_ - 1);
    const float h0 = hidden[(b0 + m) * H_ + n];
    h_s[i] = h0;
    r_s[i] = (_Float16)tanhf(h0);
  }
  wait_async0();
  __syncthreads();

  const int ncol = wave * 16;
  const int gid  = lane >> 4, nl = lane & 15;

  for (int t = 0; t < T_; ++t) {
    const int cur = t & 1, nxt = cur ^ 1;
    const float* in_c = in_raw + cur * BT * I_;
    const float* nz_c = nz_raw + cur * BT * I_;
    const float* re_c = re_raw + cur * BT * H_;

    // ---- async prefetch of step t+1 overlaps this step's GEMMs ----
    if (t + 1 < T_)
      prefetch_step(t + 1, b0, wave, lane, inputs, in_noise, re_noise,
                    in_raw + nxt * BT * I_, nz_raw + nxt * BT * I_, re_raw + nxt * BT * H_);

    // ---- x = x_t * (1 + in_noise) as f16, from LDS-resident raw buffers ----
    {
      const int e0 = tid * 4;       // BT*I = 2048 = NTH*4 -> one quad per thread
#pragma unroll
      for (int j = 0; j < 4; ++j)
        x_s[e0 + j] = (_Float16)(in_c[e0 + j] * (1.0f + nz_c[e0 + j]));
    }
    __syncthreads();

    // ---- acc = tanh(h) @ whh^T + x @ W_ih^T  (one 16x16 tile per wave) ----
    v8f acc = {};
#pragma unroll
    for (int kk = 0; kk < H_; kk += 32) {
      v16h a = load_matA(r_s, H_, kk, lane);
      v16h b = load_matB(whh_s, H_, ncol, kk, lane);
      acc = __builtin_amdgcn_wmma_f32_16x16x32_f16(false, a, false, b, (short)0, acc, false, false);
    }
#pragma unroll
    for (int kk = 0; kk < I_; kk += 32) {
      v16h a = load_matA(x_s, I_, kk, lane);
      v16h b = load_matB(wih_s, I_, ncol, kk, lane);
      acc = __builtin_amdgcn_wmma_f32_16x16x32_f16(false, a, false, b, (short)0, acc, false, false);
    }
    __syncthreads();   // all r_s reads finished before overwrite

    // ---- leaky-integrator update in WMMA C/D layout; refresh h_s, r_s ----
    {
      const int n = ncol + nl;
      const float bn = bias_s[n];
#pragma unroll
      for (int v = 0; v < 8; ++v) {
        const int m = v + 8 * gid;
        const float ho = h_s[m * H_ + n];
        const float hn = ho + ALPHA * (-ho + acc[v] + bn + re_c[m * H_ + n]);
        h_s[m * H_ + n] = hn;
        r_s[m * H_ + n] = (_Float16)tanhf(hn);
      }
    }
    wait_async0();     // t+1 raw buffers resident before anyone can read them
    __syncthreads();

    // ---- out = tanh(h_new) @ W_ho^T + b_ho  (O=64 -> waves 0..3) ----
    if (wave < 4) {
      const int oc = wave * 16;
      v8f occ = {};
#pragma unroll
      for (int kk = 0; kk < H_; kk += 32) {
        v16h a = load_matA(r_s, H_, kk, lane);
        v16h b = load_matB(who_s, H_, oc, kk, lane);
        occ = __builtin_amdgcn_wmma_f32_16x16x32_f16(false, a, false, b, (short)0, occ, false, false);
      }
      const int n = oc + nl;
      const float bo = bho_s[n];
#pragma unroll
      for (int v = 0; v < 8; ++v) {
        const int m = v + 8 * gid;
        outs[(long)(b0 + m) * T_ * O_ + (long)t * O_ + n] = occ[v] + bo;
      }
    }
  }

  __syncthreads();
  for (int i = tid; i < BT * H_; i += NTH) {
    const int m = i >> 8, n = i & (H_ - 1);
    h_final[(long)(b0 + m) * H_ + n] = h_s[i];
  }
}

extern "C" void kernel_launch(void* const* d_in, const int* in_sizes, int n_in,
                              void* d_out, int out_size, void* d_ws, size_t ws_size,
                              hipStream_t stream) {
  (void)in_sizes; (void)n_in; (void)d_ws; (void)ws_size; (void)out_size;
  const float* inputs   = (const float*)d_in[0];
  const float* hidden   = (const float*)d_in[1];
  const float* hh       = (const float*)d_in[2];
  const float* mask     = (const float*)d_in[3];
  const float* W_ih     = (const float*)d_in[4];
  const float* bias     = (const float*)d_in[5];
  const float* W_ho     = (const float*)d_in[6];
  const float* b_ho     = (const float*)d_in[7];
  const float* in_noise = (const float*)d_in[8];
  const float* re_noise = (const float*)d_in[9];

  float* h_final = (float*)d_out;               // [B,H]
  float* outs    = h_final + B_ * H_;           // [B,T,O]

  const size_t smem =
      (size_t)H_ * H_ * sizeof(_Float16) +      // whh
      (size_t)H_ * I_ * sizeof(_Float16) +      // W_ih
      (size_t)O_ * H_ * sizeof(_Float16) +      // W_ho
      (size_t)BT * H_ * sizeof(float)   +       // h
      (size_t)BT * H_ * sizeof(_Float16) +      // tanh(h)
      (size_t)BT * I_ * sizeof(_Float16) +      // x
      (size_t)H_ * sizeof(float) +              // bias
      (size_t)O_ * sizeof(float) +              // b_ho
      (size_t)2 * BT * I_ * sizeof(float) +     // inputs raw, double-buffered
      (size_t)2 * BT * I_ * sizeof(float) +     // input_noise raw, double-buffered
      (size_t)2 * BT * H_ * sizeof(float);      // re_noise raw, double-buffered
      // = 324,864 B -> 159 x 2KB LDS blocks = 325,632 B <= 320 KB/WGP

  // Raise dynamic-LDS cap (idempotent; not a stream op, graph-capture safe).
  (void)hipFuncSetAttribute((const void*)ei_rnn_kernel,
                            hipFuncAttributeMaxDynamicSharedMemorySize, (int)smem);

  ei_rnn_kernel<<<B_ / BT, NTH, smem, stream>>>(
      inputs, hidden, hh, mask, W_ih, bias, W_ho, b_ho, in_noise, re_noise,
      h_final, outs);
}
